// MultiHeadAttention_70970039599825
// MI455X (gfx1250) — matrix operand order repaired and checked by reading, not compiled, not run on hardware
//
#include <hip/hip_runtime.h>
#include <hip/hip_bf16.h>

// Shapes from the reference: B=2, S=2048, D=1024, H=16, HS=64
#define BATCH 2
#define SEQ   2048
#define DIM   1024
#define NH    16
#define HSZ   64

typedef __attribute__((ext_vector_type(16))) __bf16 v16bf;
typedef __attribute__((ext_vector_type(8)))  __bf16 v8bf;
typedef __attribute__((ext_vector_type(4)))  __bf16 v4bf;
typedef __attribute__((ext_vector_type(8)))  float  v8f;
typedef __attribute__((ext_vector_type(4)))  float  v4f;

static __device__ inline __bf16 f2bf(float x) { return (__bf16)x; }

static __device__ inline v8f zero_v8f() {
  v8f z = {0.f, 0.f, 0.f, 0.f, 0.f, 0.f, 0.f, 0.f};
  return z;
}

// A-matrix 16-bit fragment (16x32, one WMMA K step):
// lane half lh: K = lh*16 + [0..15] contiguous
static __device__ inline v16bf load_a_frag(const __bf16* row, int lh) {
  union { v16bf v; v8bf h[2]; } u;
  u.h[0] = *reinterpret_cast<const v8bf*>(row + lh * 16);
  u.h[1] = *reinterpret_cast<const v8bf*>(row + lh * 16 + 8);
  return u.v;
}

// B-matrix 16-bit fragment (32x16): lane = column, K = lh*16 + [0..15] contiguous.
// `col16` must already point at the 16 contiguous K values for this lane.
static __device__ inline v16bf load_b_frag(const __bf16* col16) {
  union { v16bf v; v8bf h[2]; } u;
  u.h[0] = *reinterpret_cast<const v8bf*>(col16);
  u.h[1] = *reinterpret_cast<const v8bf*>(col16 + 8);
  return u.v;
}

// CDNA5 async global->LDS copy (16B per lane), tracked by ASYNCcnt.
// Generic __shared__ pointers: low 32 bits are the LDS byte offset (ISA 10.2).
static __device__ inline void async_copy_b128(void* lds_dst, const __bf16* gsrc) {
  const unsigned lds_addr = (unsigned)(uintptr_t)lds_dst;
  asm volatile("global_load_async_to_lds_b128 %0, %1, off"
               :
               : "v"(lds_addr), "v"(gsrc)
               : "memory");
}
static __device__ inline void wait_async_zero() {
  asm volatile("s_wait_asynccnt 0" ::: "memory");
}

// ---------------- prep kernels (one-shot, tiny vs GEMM cost) ----------------

// f32 -> bf16, 8 elements per thread
__global__ __launch_bounds__(256)
void convert_bf16(const float* __restrict__ in, __bf16* __restrict__ out) {
  const size_t i = ((size_t)blockIdx.x * 256 + threadIdx.x) * 8;
  v4f a = *reinterpret_cast<const v4f*>(in + i);
  v4f b = *reinterpret_cast<const v4f*>(in + i + 4);
  v8bf o;
#pragma unroll
  for (int e = 0; e < 4; ++e) { o[e] = f2bf(a[e]); o[e + 4] = f2bf(b[e]); }
  *reinterpret_cast<v8bf*>(out + i) = o;
}

// W[k][n] f32 -> Wt[n][k] bf16, 64x64 tiles through LDS
__global__ __launch_bounds__(256)
void transpose_bf16(const float* __restrict__ W, __bf16* __restrict__ Wt) {
  __shared__ __bf16 t[64][72];  // pad: stride 144B
  const int tid = threadIdx.x;
  const int k0 = blockIdx.x * 64;
  const int n0 = blockIdx.y * 64;
  {
    const int r = tid >> 4;          // 0..15
    const int c = (tid & 15) * 4;    // 0..60
#pragma unroll
    for (int it = 0; it < 4; ++it) {
      const int row = r + it * 16;
      v4f v = *reinterpret_cast<const v4f*>(W + (size_t)(k0 + row) * DIM + n0 + c);
#pragma unroll
      for (int e = 0; e < 4; ++e) t[row][c + e] = f2bf(v[e]);
    }
  }
  __syncthreads();
  {
    const int nr  = tid >> 2;        // 0..63
    const int seg = (tid & 3) * 16;  // 0,16,32,48
    union { v16bf v; __bf16 e[16]; } u;
#pragma unroll
    for (int e = 0; e < 16; ++e) u.e[e] = t[seg + e][nr];
    *reinterpret_cast<v16bf*>(Wt + (size_t)(n0 + nr) * DIM + k0 + seg) = u.v;
  }
}

// ---------------- main GEMM: C[M x 1024] = A * Wt^T + bias ----------------
// A: bf16 [M,1024] row-major. Wt: bf16 [1024 n][1024 k] (pre-transposed weight).
// mode 0: bf16 out row-major; mode 1: bf16 out transposed per batch (V^T);
// mode 2: f32 out row-major.
// LDS-free: both operands fragment-loaded from global; working set is L2-resident
// (192MB L2) and each wave frag load consumes 16 rows x 64B of cacheline fully.
__global__ __launch_bounds__(256)
void gemm_bf16(const __bf16* __restrict__ A, const __bf16* __restrict__ Wt,
               const float* __restrict__ bias, void* __restrict__ Out, int mode) {
  const int tid  = threadIdx.x;
  const int lane = tid & 31;
  const int wid  = tid >> 5;
  const int wm   = wid >> 2;      // 0..1 -> 64-row slab
  const int wn   = wid & 3;       // 0..3 -> 32-col slab
  const int lh   = lane >> 4;     // lane half
  const int l16  = lane & 15;
  const int m0   = blockIdx.y * 128;
  const int n0   = blockIdx.x * 128;

  const __bf16* Ab = A  + (size_t)(m0 + wm * 64 + l16) * DIM;
  const __bf16* Bb = Wt + (size_t)(n0 + wn * 32 + l16) * DIM + lh * 16;

  v8f acc[4][2];
#pragma unroll
  for (int i = 0; i < 4; ++i)
#pragma unroll
    for (int j = 0; j < 2; ++j) acc[i][j] = zero_v8f();

#pragma unroll 2
  for (int k0 = 0; k0 < DIM; k0 += 32) {
    v16bf af[4], bfr[2];
#pragma unroll
    for (int i = 0; i < 4; ++i)
      af[i] = load_a_frag(Ab + (size_t)i * 16 * DIM + k0, lh);
#pragma unroll
    for (int j = 0; j < 2; ++j)
      bfr[j] = load_b_frag(Bb + (size_t)j * 16 * DIM + k0);
#pragma unroll
    for (int i = 0; i < 4; ++i)
#pragma unroll
      for (int j = 0; j < 2; ++j)
        acc[i][j] = __builtin_amdgcn_wmma_f32_16x16x32_bf16(
            false, af[i], false, bfr[j], (short)0, acc[i][j], false, false);
    if (k0 + 64 < DIM) {
      __builtin_prefetch(Ab + k0 + 64, 0, 1);
      __builtin_prefetch(Bb + k0 + 64, 0, 1);
    }
  }

  // --- epilogue ---
#pragma unroll
  for (int i = 0; i < 4; ++i) {
    const int tm = m0 + wm * 64 + i * 16;
    const int mbase = tm + 8 * lh;  // rows mbase..mbase+7 in VGPRs 0..7
#pragma unroll
    for (int j = 0; j < 2; ++j) {
      const int n = n0 + wn * 32 + j * 16 + l16;
      const float bn = bias[n];
      if (mode == 1) {
        const int bb = mbase >> 11, s = mbase & 2047;
        v8bf pk;
#pragma unroll
        for (int r = 0; r < 8; ++r) pk[r] = f2bf(acc[i][j][r] + bn);
        *reinterpret_cast<v8bf*>((__bf16*)Out + ((size_t)(bb * 1024 + n)) * 2048 + s) = pk;
      } else if (mode == 0) {
#pragma unroll
        for (int r = 0; r < 8; ++r)
          ((__bf16*)Out)[(size_t)(mbase + r) * DIM + n] = f2bf(acc[i][j][r] + bn);
      } else {
#pragma unroll
        for (int r = 0; r < 8; ++r)
          ((float*)Out)[(size_t)(mbase + r) * DIM + n] = acc[i][j][r] + bn;
      }
    }
  }
}

// Flash attention, causal. Q,K: bf16 [B*S, D]. Vt: bf16 [B][D][S]. O: bf16 [B*S, D].
// Block = 8 waves = 128 query rows of one (b,h). K/V^T KV-blocks (64 columns) are
// staged once per block into LDS with async global->LDS copies (ASYNCcnt), then
// all 8 waves fragment-load them from LDS: 8x less global traffic than per-wave
// fetches. Waves iterate the block-uniform causal range; out-of-range waves skip
// compute (wave-uniform branch, EXEC stays all-ones for WMMA) but keep barriers.
__global__ __launch_bounds__(256)
void attn_flash(const __bf16* __restrict__ Q, const __bf16* __restrict__ K,
                const __bf16* __restrict__ Vt, __bf16* __restrict__ O) {
  __shared__ __bf16 ks[64][72];      // K block   [kv][hs], stride 144B
  __shared__ __bf16 vs[64][72];      // V^T block [hs][kv], stride 144B
  __shared__ __bf16 pst[8][16][80];  // per-wave P staging, stride 160B

  const int tid  = threadIdx.x;
  const int lane = tid & 31;
  const int w    = tid >> 5;
  const int lh   = lane >> 4;
  const int l16  = lane & 15;

  const int qblk = blockIdx.x & 15;        // S/128 = 16
  const int bh   = blockIdx.x >> 4;
  const int b    = bh >> 4;                // /H
  const int h    = bh & 15;
  const int q0   = qblk * 128 + w * 16;

  const size_t rowbase = (size_t)b * SEQ * DIM;
  const int hc = h * HSZ;

  // Q fragments (A operand), loaded once
  v16bf aq[2];
#pragma unroll
  for (int kk = 0; kk < 2; ++kk)
    aq[kk] = load_a_frag(Q + rowbase + (size_t)(q0 + l16) * DIM + hc + kk * 32, lh);

  v8f o_acc[4];
#pragma unroll
  for (int jh = 0; jh < 4; ++jh) o_acc[jh] = zero_v8f();
  float mi[8], li[8];
#pragma unroll
  for (int r = 0; r < 8; ++r) { mi[r] = -3.0e38f; li[r] = 0.f; }

  // block-uniform causal range: max kv needed by any wave in this block
  const int nblk = 2 * qblk + 2;  // (qblk*128 + 128) / 64
  for (int t = 0; t < nblk; ++t) {
    const int kv0 = t * 64;

    // --- cooperative async stage of K and V^T tiles (8KB each) ---
    {
      const int r = tid >> 2;          // 0..63
      const int c = (tid & 3) * 16;    // elements: 0,16,32,48
      const __bf16* gk = K + rowbase + (size_t)(kv0 + r) * DIM + hc + c;
      const __bf16* gv = Vt + ((size_t)(b * DIM + hc + r)) * SEQ + kv0 + c;
      async_copy_b128(&ks[r][c], gk);
      async_copy_b128(&ks[r][c + 8], gk + 8);
      async_copy_b128(&vs[r][c], gv);
      async_copy_b128(&vs[r][c + 8], gv + 8);
    }
    wait_async_zero();
    __syncthreads();

    if (kv0 <= q0 + 15) {  // wave-uniform: this wave still has causal work
      // S = Q * K^T  (4 tiles of 16 kv columns), B-frags from LDS
      v8f sc[4];
#pragma unroll
      for (int j = 0; j < 4; ++j) {
        v8f c = zero_v8f();
#pragma unroll
        for (int kk = 0; kk < 2; ++kk) {
          v16bf bk = load_b_frag(&ks[j * 16 + l16][kk * 32 + lh * 16]);
          c = __builtin_amdgcn_wmma_f32_16x16x32_bf16(false, aq[kk], false, bk,
                                                      (short)0, c, false, false);
        }
        sc[j] = c;
      }

      // scale + causal mask
#pragma unroll
      for (int j = 0; j < 4; ++j) {
        const int n = kv0 + j * 16 + l16;
#pragma unroll
        for (int r = 0; r < 8; ++r) {
          const int m = q0 + r + 8 * lh;
          const float s = sc[j][r] * 0.125f;  // 1/sqrt(64)
          sc[j][r] = (n > m) ? -3.0e38f : s;
        }
      }

      // online softmax (row = r + 8*lh spans the 16 lanes of this half)
#pragma unroll
      for (int r = 0; r < 8; ++r) {
        float v = fmaxf(fmaxf(sc[0][r], sc[1][r]), fmaxf(sc[2][r], sc[3][r]));
#pragma unroll
        for (int off = 1; off < 16; off <<= 1) v = fmaxf(v, __shfl_xor(v, off, 32));
        const float mnew = fmaxf(mi[r], v);
        const float corr = __expf(mi[r] - mnew);
        float rs = 0.f;
#pragma unroll
        for (int j = 0; j < 4; ++j) {
          const float p = __expf(sc[j][r] - mnew);
          sc[j][r] = p;
          rs += p;
        }
#pragma unroll
        for (int off = 1; off < 16; off <<= 1) rs += __shfl_xor(rs, off, 32);
        li[r] = li[r] * corr + rs;
        mi[r] = mnew;
#pragma unroll
        for (int jh = 0; jh < 4; ++jh) o_acc[jh][r] *= corr;
      }

      // stage P (C-layout -> LDS row-major), then reload in A-layout.
      // DS ops are in-order within a wave; buffer is wave-private.
#pragma unroll
      for (int j = 0; j < 4; ++j)
#pragma unroll
        for (int r = 0; r < 8; ++r)
          pst[w][r + 8 * lh][j * 16 + l16] = f2bf(sc[j][r]);

      v16bf ap[2];
#pragma unroll
      for (int kk = 0; kk < 2; ++kk)
        ap[kk] = load_a_frag(&pst[w][l16][kk * 32], lh);

      // O += P * V, B-frags from LDS (V^T rows contiguous in kv)
#pragma unroll
      for (int jh = 0; jh < 4; ++jh) {
#pragma unroll
        for (int kk = 0; kk < 2; ++kk) {
          v16bf bv = load_b_frag(&vs[jh * 16 + l16][kk * 32 + lh * 16]);
          o_acc[jh] = __builtin_amdgcn_wmma_f32_16x16x32_bf16(
              false, ap[kk], false, bv, (short)0, o_acc[jh], false, false);
        }
      }
    }
    __syncthreads();  // protect ks/vs before next iteration's async stage
  }

  // normalize + store O (bf16, [B*S, D])
#pragma unroll
  for (int jh = 0; jh < 4; ++jh) {
    const int n = hc + jh * 16 + l16;
#pragma unroll
    for (int r = 0; r < 8; ++r) {
      const int m = q0 + r + 8 * lh;
      O[rowbase + (size_t)m * DIM + n] = f2bf(o_acc[jh][r] / li[r]);
    }
  }
}

extern "C" void kernel_launch(void* const* d_in, const int* in_sizes, int n_in,
                              void* d_out, int out_size, void* d_ws, size_t ws_size,
                              hipStream_t stream) {
  (void)in_sizes; (void)n_in; (void)out_size; (void)ws_size;
  const float* q  = (const float*)d_in[0];
  const float* Wq = (const float*)d_in[1];
  const float* bq = (const float*)d_in[2];
  const float* Wk = (const float*)d_in[3];
  const float* bk = (const float*)d_in[4];
  const float* Wv = (const float*)d_in[5];
  const float* bv = (const float*)d_in[6];
  const float* Wo = (const float*)d_in[7];
  const float* bo = (const float*)d_in[8];

  const size_t act = (size_t)BATCH * SEQ * DIM;  // 4M elements
  const size_t wel = (size_t)DIM * DIM;          // 1M elements
  __bf16* ab  = (__bf16*)d_ws;     // q bf16 [B*S, D]
  __bf16* wqt = ab + act;          // Wq^T bf16 [n][k]
  __bf16* wkt = wqt + wel;
  __bf16* wvt = wkt + wel;
  __bf16* wot = wvt + wel;
  __bf16* qb  = wot + wel;         // Q bf16 [B*S, D]
  __bf16* kb  = qb + act;          // K bf16 [B*S, D]
  __bf16* vt  = kb + act;          // V^T bf16 [B][D][S]
  __bf16* ob  = vt + act;          // attention out bf16 [B*S, D]

  // prep: one-time convert/transpose into bf16
  convert_bf16<<<act / (256 * 8), 256, 0, stream>>>(q, ab);
  dim3 tg(DIM / 64, DIM / 64);  // (16,16)
  transpose_bf16<<<tg, 256, 0, stream>>>(Wq, wqt);
  transpose_bf16<<<tg, 256, 0, stream>>>(Wk, wkt);
  transpose_bf16<<<tg, 256, 0, stream>>>(Wv, wvt);
  transpose_bf16<<<tg, 256, 0, stream>>>(Wo, wot);

  dim3 gg(DIM / 128, (BATCH * SEQ) / 128);  // (8, 32)
  gemm_bf16<<<gg, 256, 0, stream>>>(ab, wqt, bq, (void*)qb, 0);
  gemm_bf16<<<gg, 256, 0, stream>>>(ab, wkt, bk, (void*)kb, 0);
  gemm_bf16<<<gg, 256, 0, stream>>>(ab, wvt, bv, (void*)vt, 1);

  attn_flash<<<BATCH * NH * (SEQ / 128), 256, 0, stream>>>(qb, kb, vt, ob);

  gemm_bf16<<<gg, 256, 0, stream>>>(ob, wot, bo, d_out, 2);
}
